// LRUOptimizedHalf_80161269613386
// MI455X (gfx1250) — matrix operand order repaired and checked by reading, not compiled, not run on hardware
//
#include <hip/hip_runtime.h>
#include <hip/hip_bf16.h>

// ---------------- problem constants ----------------
constexpr int kB   = 16;      // batch
constexpr int kT   = 8192;    // time
constexpr int kIn  = 128;     // input dim
constexpr int kOut = 128;     // output dim
constexpr int kH   = 256;     // hidden dim
constexpr int kC   = 16;      // scan chunks
constexpr int kL   = kT / kC; // 512 (2^9)

constexpr int kPadIn = kIn + 8;      // 136 fp16: 272B row -> 4-bank rotation/row

typedef _Float16 v16h __attribute__((ext_vector_type(16)));
typedef _Float16 v8h  __attribute__((ext_vector_type(8)));
typedef _Float16 v2h  __attribute__((ext_vector_type(2)));
typedef float    v8f  __attribute__((ext_vector_type(8)));
typedef float    v4f  __attribute__((ext_vector_type(4)));
typedef int      v4i  __attribute__((vector_size(16)));

// pointer types matching the async-to-LDS builtin's expected parameters
typedef __attribute__((address_space(1))) v4i* lru_gptr_t;  // global int4*
typedef __attribute__((address_space(3))) v4i* lru_lptr_t;  // LDS int4*

// ---------- CDNA5 async global->LDS staging (guarded), fallback = ds stores ----------
#if defined(__HIP_DEVICE_COMPILE__) && __has_builtin(__builtin_amdgcn_global_load_async_to_lds_b128)
#define LRU_ASYNC_LDS 1
#endif

__device__ inline void lru_cp16(_Float16* dst_lds, const _Float16* src_glob) {
#ifdef LRU_ASYNC_LDS
  __builtin_amdgcn_global_load_async_to_lds_b128(
      (lru_gptr_t)src_glob,
      (lru_lptr_t)dst_lds,
      0, 0);
#else
  *(v8h*)dst_lds = *(const v8h*)src_glob;
#endif
}

__device__ inline void lru_cp_wait() {
#ifdef LRU_ASYNC_LDS
#if __has_builtin(__builtin_amdgcn_s_wait_asynccnt)
  __builtin_amdgcn_s_wait_asynccnt(0);
#else
  asm volatile("s_wait_asynccnt 0" ::: "memory");
#endif
#endif
  __syncthreads();
}

// Build WMMA A-operand (16x32 fp16) from an fp32 row, converting on the fly.
// lane<16 -> row=lane, K halves {0..7,16..23}; lane>=16 -> row=lane-16, K halves {8..15,24..31}.
__device__ inline v16h lru_cvt_a_f32(const float* __restrict__ xrow, int kb) {
  v4f f0 = *(const v4f*)(xrow + kb);
  v4f f1 = *(const v4f*)(xrow + kb + 4);
  v4f f2 = *(const v4f*)(xrow + kb + 16);
  v4f f3 = *(const v4f*)(xrow + kb + 20);
  v16h a;
#pragma unroll
  for (int j = 0; j < 4; ++j) {
    a[j]      = (_Float16)f0[j];
    a[4 + j]  = (_Float16)f1[j];
    a[8 + j]  = (_Float16)f2[j];
    a[12 + j] = (_Float16)f3[j];
  }
  return a;
}

// A-operand from fp16 row (elements 0..7 at kb, 8..15 at kb+16)
__device__ inline v16h lru_ld_a16(const _Float16* __restrict__ arow, int kb) {
  v8h a0 = *(const v8h*)(arow + kb);
  v8h a1 = *(const v8h*)(arow + kb + 16);
  v16h a;
#pragma unroll
  for (int j = 0; j < 8; ++j) { a[j] = a0[j]; a[8 + j] = a1[j]; }
  return a;
}

// B-operand: elements 0..7 at p, 8..15 at p+16 (two 16B-aligned halves)
__device__ inline v16h lru_ld_b16(const _Float16* p) {
  v8h b0 = *(const v8h*)p;
  v8h b1 = *(const v8h*)(p + 16);
  v16h b;
#pragma unroll
  for (int j = 0; j < 8; ++j) { b[j] = b0[j]; b[8 + j] = b1[j]; }
  return b;
}

// ---------------- prep: constants + fp16 weight packing ----------------
__global__ void lru_prep(const float* __restrict__ nu, const float* __restrict__ th,
                         const float* __restrict__ Br, const float* __restrict__ Bi,
                         const float* __restrict__ Cr, const float* __restrict__ Ci,
                         const float* __restrict__ Dm,
                         const float* __restrict__ bhr, const float* __restrict__ bhi,
                         const float* __restrict__ bo,
                         _Float16* __restrict__ Brg, _Float16* __restrict__ Big,
                         _Float16* __restrict__ Wc,  _Float16* __restrict__ Wd,
                         float* __restrict__ LR, float* __restrict__ LI,
                         float* __restrict__ bRg, float* __restrict__ bIg,
                         float* __restrict__ bO) {
  const int tid = blockIdx.x * blockDim.x + threadIdx.x;
  const int nth = gridDim.x * blockDim.x;

  for (int h = tid; h < kH; h += nth) {
    float e_nu = __expf(nu[h]);
    float e_th = __expf(th[h]);
    float mag  = __expf(-e_nu);
    LR[h] = mag * __cosf(e_th);
    LI[h] = mag * __sinf(e_th);
    float g  = sqrtf(fmaxf(0.f, 1.f - mag * mag));
    float gh = (float)(_Float16)g;
    bRg[h] = (float)(_Float16)bhr[h] * gh;
    bIg[h] = (float)(_Float16)bhi[h] * gh;
  }
  for (int o = tid; o < kOut; o += nth) bO[o] = (float)(_Float16)bo[o];

  for (int idx = tid; idx < kH * kIn; idx += nth) {
    int h = idx / kIn;
    float e_nu = __expf(nu[h]);
    float mag  = __expf(-e_nu);
    float g    = (float)(_Float16)sqrtf(fmaxf(0.f, 1.f - mag * mag));
    Brg[idx] = (_Float16)((float)(_Float16)Br[idx] * g);
    Big[idx] = (_Float16)((float)(_Float16)Bi[idx] * g);
  }
  for (int idx = tid; idx < kOut * kH; idx += nth) {
    int o = idx / kH, h = idx % kH;
    Wc[o * (2 * kH) + 2 * h]     = (_Float16)Cr[idx];
    Wc[o * (2 * kH) + 2 * h + 1] = (_Float16)(-Ci[idx]);
  }
  for (int idx = tid; idx < kOut * kIn; idx += nth) Wd[idx] = (_Float16)Dm[idx];
}

// ---------------- GEMM 1: Xr/Xi = (gamma*B) @ X + gamma*bias -> [B,H,T] fp16 ----------------
// 256 blocks x 256 thr; block owns 32 M-tiles (wave*4+mi); weights staged to LDS in
// 64-row quarters via async global->LDS, consumed with ds_load_b128 + WMMA.
__global__ void lru_gemm_in(const float* __restrict__ X,
                            const _Float16* __restrict__ Brg, const _Float16* __restrict__ Big,
                            const float* __restrict__ bRg, const float* __restrict__ bIg,
                            _Float16* __restrict__ Xr, _Float16* __restrict__ Xi) {
  __shared__ _Float16 sW[2][64][kPadIn];   // 34.8 KB

  const int wave = threadIdx.x >> 5;
  const int lane = threadIdx.x & 31;
  const int row  = lane & 15;
  const int half = lane >> 4;
  const int hc   = lane & 15;

  for (int mi = 0; mi < 4; ++mi) {
    const int  mtile = blockIdx.x * 32 + wave * 4 + mi;
    const long m0    = (long)mtile * 16;
    const long b     = m0 / kT;
    const long t0    = m0 % kT;

    const float* xrow = X + (m0 + row) * kIn;
    v16h a[4];
#pragma unroll
    for (int kt = 0; kt < 4; ++kt) a[kt] = lru_cvt_a_f32(xrow, kt * 32 + half * 8);

    for (int q = 0; q < 4; ++q) {
      // stage 64 h-rows of Brg and Big (32 KB) into LDS
      for (int c = threadIdx.x; c < 2048; c += 256) {
        const int m   = c >> 10;         // 0: Brg, 1: Big
        const int cc  = c & 1023;
        const int r   = cc >> 4;         // row in quarter
        const int col = (cc & 15) << 3;  // fp16 column (16B chunks)
        const _Float16* src = (m ? Big : Brg) + (size_t)(q * 64 + r) * kIn + col;
        lru_cp16(&sW[m][r][col], src);
      }
      lru_cp_wait();

#pragma unroll
      for (int ntl = 0; ntl < 4; ++ntl) {
        const int hl = ntl * 16 + hc;    // row in staged quarter
        const int h  = q * 64 + hl;
        v8f accr = {}, acci = {};
#pragma unroll
        for (int kt = 0; kt < 4; ++kt) {
          const int koff = kt * 32 + half * 8;
          v16h br = lru_ld_b16(&sW[0][hl][koff]);
          v16h bi = lru_ld_b16(&sW[1][hl][koff]);
          accr = __builtin_amdgcn_wmma_f32_16x16x32_f16(false, a[kt], false, br, (short)0, accr, false, false);
          acci = __builtin_amdgcn_wmma_f32_16x16x32_f16(false, a[kt], false, bi, (short)0, acci, false, false);
        }
        const float cr = bRg[h], ci = bIg[h];
        v8h outr, outi;
#pragma unroll
        for (int r2 = 0; r2 < 8; ++r2) {
          outr[r2] = (_Float16)(accr[r2] + cr);
          outi[r2] = (_Float16)(acci[r2] + ci);
        }
        const long base = ((long)b * kH + h) * kT + t0 + half * 8;
        *(v8h*)(Xr + base) = outr;   // 128-bit store, 8 consecutive t
        *(v8h*)(Xi + base) = outi;
      }
      __syncthreads();   // before restaging sW
    }
  }
}

// ---------------- scan pass 1: per-chunk zero-init final states ----------------
__global__ void lru_scan_state(const _Float16* __restrict__ Xr, const _Float16* __restrict__ Xi,
                               const float* __restrict__ LR, const float* __restrict__ LI,
                               float2* __restrict__ state) {
  const int h = threadIdx.x;
  const int b = blockIdx.x / kC;
  const int c = blockIdx.x % kC;
  const float lr = LR[h], li = LI[h];
  const long base = ((long)b * kH + h) * kT + (long)c * kL;
  const _Float16* xr = Xr + base;
  const _Float16* xi = Xi + base;
  float hr = 0.f, hi = 0.f;
  for (int t0 = 0; t0 < kL; t0 += 8) {
    __builtin_prefetch(xr + t0 + 64, 0, 1);
    __builtin_prefetch(xi + t0 + 64, 0, 1);
    v8h xrv = *(const v8h*)(xr + t0);
    v8h xiv = *(const v8h*)(xi + t0);
#pragma unroll
    for (int j = 0; j < 8; ++j) {
      float nr = lr * hr - li * hi + (float)xrv[j];
      float ni = lr * hi + li * hr + (float)xiv[j];
      hr = nr; hi = ni;
    }
  }
  state[((long)b * kH + h) * kC + c] = make_float2(hr, hi);
}

// ---------------- scan pass 2: combine carries across chunks (lam^512 by squaring) --------
__global__ void lru_scan_carry(const float* __restrict__ LR, const float* __restrict__ LI,
                               const float2* __restrict__ state, float2* __restrict__ carry) {
  const int h = threadIdx.x;
  const int b = blockIdx.x;
  float ar = LR[h], ai = LI[h];
#pragma unroll
  for (int s = 0; s < 9; ++s) {  // lam^(2^9) = lam^512
    float nr = ar * ar - ai * ai;
    float ni = 2.f * ar * ai;
    ar = nr; ai = ni;
  }
  float cr = 0.f, ci = 0.f;
  const long bh = (long)b * kH + h;
  for (int c = 0; c < kC; ++c) {
    carry[bh * kC + c] = make_float2(cr, ci);
    float2 s = state[bh * kC + c];
    float nr = ar * cr - ai * ci + s.x;
    float ni = ar * ci + ai * cr + s.y;
    cr = nr; ci = ni;
  }
}

// ---------------- scan pass 3: emit Yc = [B,T,2H] fp16 interleaved (hr,hi) ----------------
__global__ void lru_scan_emit(const _Float16* __restrict__ Xr, const _Float16* __restrict__ Xi,
                              const float* __restrict__ LR, const float* __restrict__ LI,
                              const float2* __restrict__ carry, _Float16* __restrict__ Yc) {
  const int h = threadIdx.x;
  const int b = blockIdx.x / kC;
  const int c = blockIdx.x % kC;
  const float lr = LR[h], li = LI[h];
  const long base = ((long)b * kH + h) * kT + (long)c * kL;
  const _Float16* xr = Xr + base;
  const _Float16* xi = Xi + base;
  float2 f = carry[((long)b * kH + h) * kC + c];
  float hr = f.x, hi = f.y;
  _Float16* y = Yc + ((long)b * kT + (long)c * kL) * (2 * kH) + 2 * h;
  for (int t0 = 0; t0 < kL; t0 += 8) {
    __builtin_prefetch(xr + t0 + 64, 0, 1);
    __builtin_prefetch(xi + t0 + 64, 0, 1);
    v8h xrv = *(const v8h*)(xr + t0);
    v8h xiv = *(const v8h*)(xi + t0);
#pragma unroll
    for (int j = 0; j < 8; ++j) {
      float nr = lr * hr - li * hi + (float)xrv[j];
      float ni = lr * hi + li * hr + (float)xiv[j];
      hr = nr; hi = ni;
      v2h o; o[0] = (_Float16)nr; o[1] = (_Float16)ni;
      *(v2h*)(y + (long)(t0 + j) * (2 * kH)) = o;   // coalesced across h
    }
  }
}

// ---------------- GEMM 2: out = Wc(K=512) @ Yc + Wd(K=128) @ X + bias -> fp32 -----------
// 2048 blocks x 256 thr; wave = N-tile, 4 M-tiles per block; Wc staged to LDS in
// 128-column K-quarters, Wd staged once; accumulators persist across stages.
__global__ void lru_gemm_out(const _Float16* __restrict__ Yc, const float* __restrict__ X,
                             const _Float16* __restrict__ Wc, const _Float16* __restrict__ Wd,
                             const float* __restrict__ bO, float* __restrict__ out) {
  __shared__ _Float16 sW[128][kPadIn];   // 34.8 KB

  const int wave = threadIdx.x >> 5;
  const int lane = threadIdx.x & 31;
  const int row  = lane & 15;
  const int half = lane >> 4;
  const int o    = wave * 16 + (lane & 15);   // output column, also LDS row

  v8f acc[4] = {};

  // --- K = 512 over interleaved hidden state, staged in 4 quarters of 128 ---
  for (int q = 0; q < 4; ++q) {
    for (int c = threadIdx.x; c < 2048; c += 256) {   // 128 rows x 16 x 16B
      const int r   = c >> 4;
      const int col = (c & 15) << 3;
      lru_cp16(&sW[r][col], Wc + (size_t)r * (2 * kH) + q * 128 + col);
    }
    lru_cp_wait();

#pragma unroll
    for (int mi = 0; mi < 4; ++mi) {
      const long m0 = ((long)blockIdx.x * 4 + mi) * 16;
      const _Float16* arow = Yc + (m0 + row) * (2 * kH) + q * 128;
#pragma unroll
      for (int j = 0; j < 4; ++j) {
        v16h a = lru_ld_a16(arow, j * 32 + half * 8);
        v16h w = lru_ld_b16(&sW[o][j * 32 + half * 8]);
        acc[mi] = __builtin_amdgcn_wmma_f32_16x16x32_f16(false, a, false, w, (short)0, acc[mi], false, false);
      }
    }
    __syncthreads();
  }

  // --- K = 128 over fp32 X (D path) ---
  for (int c = threadIdx.x; c < 2048; c += 256) {
    const int r   = c >> 4;
    const int col = (c & 15) << 3;
    lru_cp16(&sW[r][col], Wd + (size_t)r * kIn + col);
  }
  lru_cp_wait();

#pragma unroll
  for (int mi = 0; mi < 4; ++mi) {
    const long m0 = ((long)blockIdx.x * 4 + mi) * 16;
    const float* xrow = X + (m0 + row) * kIn;
#pragma unroll
    for (int j = 0; j < 4; ++j) {
      v16h a = lru_cvt_a_f32(xrow, j * 32 + half * 8);
      v16h w = lru_ld_b16(&sW[o][j * 32 + half * 8]);
      acc[mi] = __builtin_amdgcn_wmma_f32_16x16x32_f16(false, a, false, w, (short)0, acc[mi], false, false);
    }
  }

  const float bo = bO[o];
#pragma unroll
  for (int mi = 0; mi < 4; ++mi) {
    const long m0 = ((long)blockIdx.x * 4 + mi) * 16;
#pragma unroll
    for (int r2 = 0; r2 < 8; ++r2) {
      const long m = m0 + r2 + half * 8;
      out[m * kOut + o] = acc[mi][r2] + bo;
    }
  }
}

// ---------------- host launcher ----------------
extern "C" void kernel_launch(void* const* d_in, const int* in_sizes, int n_in,
                              void* d_out, int out_size, void* d_ws, size_t ws_size,
                              hipStream_t stream) {
  (void)in_sizes; (void)n_in; (void)out_size; (void)ws_size;

  const float* X   = (const float*)d_in[0];
  const float* nu  = (const float*)d_in[1];
  const float* th  = (const float*)d_in[2];
  const float* Br  = (const float*)d_in[3];
  const float* Bi  = (const float*)d_in[4];
  const float* Cr  = (const float*)d_in[5];
  const float* Ci  = (const float*)d_in[6];
  const float* Dm  = (const float*)d_in[7];
  const float* bhr = (const float*)d_in[8];
  const float* bhi = (const float*)d_in[9];
  const float* bo  = (const float*)d_in[10];

  char* ws = (char*)d_ws;
  size_t off = 0;
  auto take = [&](size_t bytes) -> void* {
    void* p = ws + off;
    off = (off + bytes + 255) & ~(size_t)255;
    return p;
  };

  _Float16* Brg = (_Float16*)take((size_t)kH * kIn * sizeof(_Float16));
  _Float16* Big = (_Float16*)take((size_t)kH * kIn * sizeof(_Float16));
  _Float16* Wc  = (_Float16*)take((size_t)kOut * 2 * kH * sizeof(_Float16));
  _Float16* Wd  = (_Float16*)take((size_t)kOut * kIn * sizeof(_Float16));
  float*    LR  = (float*)take((size_t)kH * sizeof(float));
  float*    LI  = (float*)take((size_t)kH * sizeof(float));
  float*    bRg = (float*)take((size_t)kH * sizeof(float));
  float*    bIg = (float*)take((size_t)kH * sizeof(float));
  float*    bOw = (float*)take((size_t)kOut * sizeof(float));
  float2*   st  = (float2*)take((size_t)kB * kH * kC * sizeof(float2));
  float2*   cy  = (float2*)take((size_t)kB * kH * kC * sizeof(float2));
  _Float16* Xr  = (_Float16*)take((size_t)kB * kH * kT * sizeof(_Float16));
  _Float16* Xi  = (_Float16*)take((size_t)kB * kH * kT * sizeof(_Float16));
  _Float16* Yc  = (_Float16*)take((size_t)kB * kT * 2 * kH * sizeof(_Float16));

  float* out = (float*)d_out;

  lru_prep<<<64, 256, 0, stream>>>(nu, th, Br, Bi, Cr, Ci, Dm, bhr, bhi, bo,
                                   Brg, Big, Wc, Wd, LR, LI, bRg, bIg, bOw);

  const int mtiles = (kB * kT) / 16;                 // 8192
  lru_gemm_in<<<mtiles / 32, 256, 0, stream>>>(X, Brg, Big, bRg, bIg, Xr, Xi);

  lru_scan_state<<<kB * kC, 256, 0, stream>>>(Xr, Xi, LR, LI, st);
  lru_scan_carry<<<kB, 256, 0, stream>>>(LR, LI, st, cy);
  lru_scan_emit<<<kB * kC, 256, 0, stream>>>(Xr, Xi, LR, LI, cy, Yc);

  lru_gemm_out<<<mtiles / 4, 256, 0, stream>>>(Yc, X, Wc, Wd, bOw, out);
}
